// FasterRCNN_87110526698176
// MI455X (gfx1250) — matrix operand order repaired and checked by reading, not compiled, not run on hardware
//
#include <hip/hip_runtime.h>
#include <hip/hip_bf16.h>

// ---------------- constants (all shapes are static from the reference) ----------
#define NIMG   2
#define CCH    256
#define NLVL   5
#define AANCH  3
#define NCLS   90
#define CLIN   1024
#define ROI    7
#define K9     (9*CCH)          // 2304, conv implicit-GEMM K
#define PT     80997            // total (unpadded) pixels per image across levels
#define PTP    82971            // total padded (H+2)*(W+2) pixels per image
#define FT     242991           // PT * AANCH
#define TOPK_CAP 2048
#define POOLCAP  10240
#define P2CAP    4096
#define CLIPV  4.135166556742356f   // log(1000/16)
#define KROI   (CCH*ROI*ROI)    // 12544

struct LvlInfo {
  int Hh[NLVL]; int Ww[NLVL]; int pixOff[NLVL]; int padOff[NLVL];
  int featOff[NLVL]; int Fl[NLVL];
};
struct Ptrs5 { const float* p[NLVL]; };

// ---------------- WMMA fragment types ----------------
typedef __attribute__((ext_vector_type(16))) __bf16 bfrag;
typedef __attribute__((ext_vector_type(8)))  float  v8f;
struct U32x8 { uint4 lo, hi; };

__device__ __forceinline__ bfrag make_frag(uint4 lo, uint4 hi) {
  U32x8 t{lo, hi};
  return __builtin_bit_cast(bfrag, t);
}
__device__ __forceinline__ float bf2f(unsigned short h) {
  return __uint_as_float(((unsigned)h) << 16);
}
__device__ __forceinline__ unsigned short f2bf(float f) {
  unsigned u = __float_as_uint(f);
  unsigned r = (u >> 16) & 1u;
  u += 0x7fffu + r;
  return (unsigned short)(u >> 16);
}

// async copy of 32B: global -> LDS (gfx1250 GLOBAL_LOAD_ASYNC_TO_LDS_B128, GV mode)
__device__ __forceinline__ void async_g2l_32B(unsigned ldsByteAddr, const void* gptr) {
  unsigned long long ga = (unsigned long long)(size_t)gptr;
  asm volatile("global_load_async_to_lds_b128 %0, %1, off"
               :: "v"(ldsByteAddr), "v"(ga) : "memory");
  asm volatile("global_load_async_to_lds_b128 %0, %1, off offset:16"
               :: "v"(ldsByteAddr), "v"(ga) : "memory");
}
__device__ __forceinline__ void wait_async0() {
  asm volatile("s_wait_asynccnt 0x0" ::: "memory");
}

// ---------------- utility: zero u32 buffer ----------------
__global__ void k_zero_u32(unsigned* p, size_t n) {
  size_t i = (size_t)blockIdx.x * blockDim.x + threadIdx.x;
  size_t s = (size_t)gridDim.x * blockDim.x;
  for (; i < n; i += s) p[i] = 0u;
}

// ---------------- weight / activation packing ----------------
// conv_w (Co,Ci,3,3) f32 -> Wp bf16 [co][k], k=(ky*3+kx)*256+ci
__global__ void k_pack_conv_w(const float* __restrict__ w, unsigned short* __restrict__ Wp) {
  int id = blockIdx.x * blockDim.x + threadIdx.x;
  if (id >= CCH * K9) return;
  int co = id / K9, k = id % K9;
  int tap = k / CCH, ci = k % CCH;
  int ky = tap / 3, kx = tap % 3;
  float v = w[(((size_t)co * CCH + ci) * 3 + ky) * 3 + kx];
  Wp[(size_t)co * K9 + k] = f2bf(v);
}

// fc weight (K,N) f32 row-major -> bf16 [n][k]
__global__ void k_pack_fcw(const float* __restrict__ w, unsigned short* __restrict__ Wt,
                           int K, int Nn) {
  size_t id = (size_t)blockIdx.x * blockDim.x + threadIdx.x;
  if (id >= (size_t)K * Nn) return;
  int n = (int)(id % Nn);
  int k = (int)(id / Nn);
  Wt[(size_t)n * K + k] = f2bf(w[id]);
}

// fmaps NCHW f32 -> X bf16 padded NHWC: [img][(y+1)*(W+2)+(x+1)][c]; X pre-zeroed.
__global__ void k_pack_x_pad(Ptrs5 f, LvlInfo L, unsigned short* __restrict__ X) {
  size_t id = (size_t)blockIdx.x * blockDim.x + threadIdx.x;
  size_t total = (size_t)NIMG * PT * CCH;
  if (id >= total) return;
  int c = (int)(id % CCH);
  size_t t = id / CCH;
  int gp = (int)(t % PT);
  int img = (int)(t / PT);
  int lvl = 0;
  for (int l = NLVL - 1; l >= 0; --l) if (gp >= L.pixOff[l]) { lvl = l; break; }
  int p = gp - L.pixOff[lvl];
  int Hh = L.Hh[lvl], Ww = L.Ww[lvl];
  int y = p / Ww, x = p % Ww;
  const float* fm = f.p[lvl];
  float v = fm[(((size_t)img * CCH + c) * Hh + y) * Ww + x];
  X[((size_t)img * PTP + L.padOff[lvl] + (size_t)(y + 1) * (Ww + 2) + (x + 1)) * CCH + c] = f2bf(v);
}

// ---------------- conv 3x3 + ReLU, implicit GEMM, bf16 WMMA -------------------
// Zero-halo padded input => unconditional A loads. A tile (16px x 256ch = 8KB)
// staged to LDS with async global->LDS, double-buffered across the 9 taps.
// Block 256 = 8 waves; wave w owns couts [32w,32w+32): 2 accumulators, 2 WMMA/A-frag.
__global__ void k_conv3x3_wmma(const unsigned short* __restrict__ X,
                               const unsigned short* __restrict__ Wp,
                               const float* __restrict__ bias,
                               unsigned short* __restrict__ Hout,
                               int Hh, int Ww, int pixOff, int padOff) {
  __shared__ unsigned short sA[2][16 * CCH];   // 16 KB double buffer
  const int P = Hh * Ww;
  const int Wp2 = Ww + 2;
  const int tid = threadIdx.x;
  const int lane = tid & 31;
  const int wave = tid >> 5;
  const int pixBase = blockIdx.x * 16;
  const int img = blockIdx.z;
  const int col = lane & 15;
  const int khalf = lane >> 4;

  // staging role: thread copies 32B (pixel spix, 16-elem segment sseg)
  const int spix = tid >> 4;   // 0..15
  const int sseg = tid & 15;   // 0..15
  int sp = pixBase + spix; if (sp > P - 1) sp = P - 1;
  const int syy = sp / Ww, sxx = sp % Ww;
  const unsigned short* gpix =
      X + (size_t)(img * PTP + padOff + (syy + 1) * Wp2 + (sxx + 1)) * CCH;
  const unsigned sldsBase = (unsigned)(size_t)(&sA[0][0]);
  const unsigned ldsOff = (unsigned)(spix * 512 + sseg * 32);

  // this wave's couts
  const int co0 = wave * 32 + col;
  const int co1 = co0 + 16;
  const unsigned short* wrow0 = Wp + (size_t)co0 * K9;
  const unsigned short* wrow1 = Wp + (size_t)co1 * K9;

  v8f acc0 = {}, acc1 = {};

  auto stage = [&](int buf, int tap) {
    const int dy = tap / 3 - 1, dx = tap % 3 - 1;
    const unsigned short* g = gpix + (dy * Wp2 + dx) * CCH + sseg * 16;
    async_g2l_32B(sldsBase + (unsigned)buf * (16 * CCH * 2) + ldsOff, g);
  };

  stage(0, 0);
  for (int tap = 0; tap < 9; ++tap) {
    wait_async0();
    __syncthreads();
    if (tap + 1 < 9) stage((tap + 1) & 1, tap + 1);
    const unsigned short* sa = &sA[tap & 1][0];
#pragma unroll
    for (int cb = 0; cb < 8; ++cb) {
      const uint4* ap = reinterpret_cast<const uint4*>(sa + col * CCH + cb * 32 + khalf * 16);
      uint4 a0 = ap[0], a1 = ap[1];
      const int kk = tap * 256 + cb * 32 + khalf * 16;
      const uint4* bp0 = reinterpret_cast<const uint4*>(wrow0 + kk);
      const uint4* bp1 = reinterpret_cast<const uint4*>(wrow1 + kk);
      uint4 b00 = bp0[0], b01 = bp0[1];
      uint4 b10 = bp1[0], b11 = bp1[1];
      __builtin_prefetch(wrow0 + kk + 256, 0, 1);
      bfrag af = make_frag(a0, a1);
      acc0 = __builtin_amdgcn_wmma_f32_16x16x32_bf16(
          false, af, false, make_frag(b00, b01), (short)0, acc0, false, false);
      acc1 = __builtin_amdgcn_wmma_f32_16x16x32_bf16(
          false, af, false, make_frag(b10, b11), (short)0, acc1, false, false);
    }
  }
  const float bv0 = bias[co0], bv1 = bias[co1];
  for (int r = 0; r < 8; ++r) {
    const int pp = pixBase + khalf * 8 + r;
    if (pp < P) {
      size_t o = (size_t)(img * PT + pixOff + pp) * CCH;
      Hout[o + co0] = f2bf(fmaxf(acc0[r] + bv0, 0.0f));
      Hout[o + co1] = f2bf(fmaxf(acc1[r] + bv1, 0.0f));
    }
  }
}

// ---------------- generic bf16 WMMA GEMM: Out = act(A[M,K] @ Wt[N,K]^T + b) -----
// grid (M/16, Nn/256), block 256. K % 256 == 0. A super-tile (16 x 256) staged to
// LDS via async global->LDS, double-buffered; each wave: 2 accumulators (32 couts).
__global__ void k_gemm_wmma(const unsigned short* __restrict__ A,
                            const unsigned short* __restrict__ Wt,
                            const float* __restrict__ bias,
                            unsigned short* __restrict__ Out,
                            int M, int K, int Nn, int doRelu) {
  __shared__ unsigned short sA[2][16 * 256];
  const int tid = threadIdx.x;
  const int lane = tid & 31;
  const int wave = tid >> 5;
  const int mBase = blockIdx.x * 16;
  const int col = lane & 15;
  const int khalf = lane >> 4;
  const int co0 = blockIdx.y * 256 + wave * 32 + col;
  const int co1 = co0 + 16;
  const unsigned short* wrow0 = Wt + (size_t)co0 * K;
  const unsigned short* wrow1 = Wt + (size_t)co1 * K;

  const int spix = tid >> 4, sseg = tid & 15;
  const unsigned short* arow = A + (size_t)(mBase + spix) * K;
  const unsigned sldsBase = (unsigned)(size_t)(&sA[0][0]);
  const unsigned ldsOff = (unsigned)(spix * 512 + sseg * 32);
  const int nSuper = K >> 8;

  v8f acc0 = {}, acc1 = {};

  auto stage = [&](int buf, int ss) {
    const unsigned short* g = arow + ss * 256 + sseg * 16;
    async_g2l_32B(sldsBase + (unsigned)buf * (16 * 256 * 2) + ldsOff, g);
  };

  stage(0, 0);
  for (int ss = 0; ss < nSuper; ++ss) {
    wait_async0();
    __syncthreads();
    if (ss + 1 < nSuper) stage((ss + 1) & 1, ss + 1);
    const unsigned short* sa = &sA[ss & 1][0];
#pragma unroll
    for (int cb = 0; cb < 8; ++cb) {
      const uint4* ap = reinterpret_cast<const uint4*>(sa + col * 256 + cb * 32 + khalf * 16);
      uint4 a0 = ap[0], a1 = ap[1];
      const int kk = ss * 256 + cb * 32 + khalf * 16;
      const uint4* bp0 = reinterpret_cast<const uint4*>(wrow0 + kk);
      const uint4* bp1 = reinterpret_cast<const uint4*>(wrow1 + kk);
      uint4 b00 = bp0[0], b01 = bp0[1];
      uint4 b10 = bp1[0], b11 = bp1[1];
      __builtin_prefetch(wrow0 + kk + 256, 0, 1);
      bfrag af = make_frag(a0, a1);
      acc0 = __builtin_amdgcn_wmma_f32_16x16x32_bf16(
          false, af, false, make_frag(b00, b01), (short)0, acc0, false, false);
      acc1 = __builtin_amdgcn_wmma_f32_16x16x32_bf16(
          false, af, false, make_frag(b10, b11), (short)0, acc1, false, false);
    }
  }
  const float bv0 = bias[co0], bv1 = bias[co1];
  for (int r = 0; r < 8; ++r) {
    int m = mBase + khalf * 8 + r;
    if (m < M) {
      float v0 = acc0[r] + bv0, v1 = acc1[r] + bv1;
      if (doRelu) { v0 = fmaxf(v0, 0.0f); v1 = fmaxf(v1, 0.0f); }
      Out[(size_t)m * Nn + co0] = f2bf(v0);
      Out[(size_t)m * Nn + co1] = f2bf(v1);
    }
  }
}

// ---------------- RPN heads: 3 logits + 12 reg per pixel ----------------
__global__ void k_heads(const unsigned short* __restrict__ Hf,
                        const float* __restrict__ logw, const float* __restrict__ logb,
                        const float* __restrict__ regw, const float* __restrict__ regb,
                        float* __restrict__ Lg, float* __restrict__ Rg, LvlInfo L) {
  int id = blockIdx.x * blockDim.x + threadIdx.x;
  if (id >= NIMG * PT) return;
  int img = id / PT, gp = id % PT;
  int lvl = 0;
  for (int l = NLVL - 1; l >= 0; --l) if (gp >= L.pixOff[l]) { lvl = l; break; }
  int p = gp - L.pixOff[lvl];
  const unsigned short* h = Hf + (size_t)id * CCH;
  float lacc[AANCH] = {0.f, 0.f, 0.f};
  float racc[4 * AANCH];
  for (int o = 0; o < 4 * AANCH; ++o) racc[o] = 0.f;
  for (int c = 0; c < CCH; ++c) {
    float hv = bf2f(h[c]);
    for (int a = 0; a < AANCH; ++a) lacc[a] += hv * logw[a * CCH + c];
    for (int o = 0; o < 4 * AANCH; ++o) racc[o] += hv * regw[o * CCH + c];
  }
  int fb = L.featOff[lvl] + p * AANCH;
  for (int a = 0; a < AANCH; ++a) {
    Lg[(size_t)img * FT + fb + a] = lacc[a] + logb[a];
    for (int j = 0; j < 4; ++j)
      Rg[((size_t)img * FT + fb + a) * 4 + j] = racc[a * 4 + j] + regb[a * 4 + j];
  }
}

// ---------------- per-(img,lvl) approximate top-k via histogram threshold ------
__global__ void k_topk(const float* __restrict__ Lg, LvlInfo L,
                       float* __restrict__ candScore, int* __restrict__ candIdx,
                       int* __restrict__ candCnt) {
  int slot = blockIdx.x;
  int img = slot / NLVL, lvl = slot % NLVL;
  int Fl = L.Fl[lvl];
  const float* v = Lg + (size_t)img * FT + L.featOff[lvl];
  int kWant = Fl < 1000 ? Fl : 1000;
  __shared__ float smx[256];
  __shared__ float smn[256];
  __shared__ int shist[256];
  __shared__ float sthr;
  float mx = -1e30f, mn = 1e30f;
  for (int i = threadIdx.x; i < Fl; i += 256) {
    float s = v[i];
    mx = fmaxf(mx, s);
    mn = fminf(mn, s);
  }
  smx[threadIdx.x] = mx; smn[threadIdx.x] = mn;
  __syncthreads();
  for (int o = 128; o > 0; o >>= 1) {
    if (threadIdx.x < o) {
      smx[threadIdx.x] = fmaxf(smx[threadIdx.x], smx[threadIdx.x + o]);
      smn[threadIdx.x] = fminf(smn[threadIdx.x], smn[threadIdx.x + o]);
    }
    __syncthreads();
  }
  float gmax = smx[0], gmin = smn[0];
  shist[threadIdx.x] = 0;
  __syncthreads();
  float range = gmax - gmin + 1e-6f;
  float inv = 256.0f / range;
  for (int i = threadIdx.x; i < Fl; i += 256) {
    int b = (int)((v[i] - gmin) * inv);
    if (b > 255) b = 255;
    if (b < 0) b = 0;
    atomicAdd(&shist[b], 1);
  }
  __syncthreads();
  if (threadIdx.x == 0) {
    int acc = 0, b = 255;
    for (; b >= 0; --b) { acc += shist[b]; if (acc >= kWant) break; }
    if (b < 0) b = 0;
    sthr = gmin + b * (range / 256.0f);
  }
  __syncthreads();
  float thr = sthr;
  for (int i = threadIdx.x; i < Fl; i += 256) {
    float s = v[i];
    if (s >= thr) {
      int pos = atomicAdd(&candCnt[slot], 1);
      if (pos < TOPK_CAP) {
        candScore[slot * TOPK_CAP + pos] = s;
        candIdx[slot * TOPK_CAP + pos] = i;
      }
    }
  }
}

// ---------------- decode + clamp + min-size -> per-image pool ----------------
__global__ void k_decode(const float* __restrict__ candScore, const int* __restrict__ candIdx,
                         const int* __restrict__ candCnt, const float* __restrict__ Rg,
                         Ptrs5 pri, const float* __restrict__ imsizes,
                         float* __restrict__ pool1, int* __restrict__ imgCandCnt, LvlInfo L) {
  int id = blockIdx.x * blockDim.x + threadIdx.x;
  int slot = id / TOPK_CAP;
  int j = id % TOPK_CAP;
  if (slot >= NIMG * NLVL) return;
  int cc = candCnt[slot];
  if (cc > TOPK_CAP) cc = TOPK_CAP;
  if (j >= cc) return;
  int img = slot / NLVL, lvl = slot % NLVL;
  int fi = candIdx[slot * TOPK_CAP + j];
  float logit = candScore[slot * TOPK_CAP + j];
  const float* pr = pri.p[lvl] + (size_t)fi * 4;
  const float* rg = Rg + ((size_t)img * FT + L.featOff[lvl] + fi) * 4;
  float cx = pr[0] + rg[0] * pr[2];
  float cy = pr[1] + rg[1] * pr[3];
  float w = pr[2] * __expf(fminf(rg[2], CLIPV));
  float h = pr[3] * __expf(fminf(rg[3], CLIPV));
  float Wi = imsizes[img * 2 + 1], Hi = imsizes[img * 2 + 0];
  float x1 = fminf(fmaxf(cx - w * 0.5f, 0.f), Wi);
  float x2 = fminf(fmaxf(cx + w * 0.5f, 0.f), Wi);
  float y1 = fminf(fmaxf(cy - h * 0.5f, 0.f), Hi);
  float y2 = fminf(fmaxf(cy + h * 0.5f, 0.f), Hi);
  if ((x2 - x1) >= 1e-3f && (y2 - y1) >= 1e-3f) {
    int pos = atomicAdd(&imgCandCnt[img], 1);
    if (pos < POOLCAP) {
      float* e = pool1 + ((size_t)img * POOLCAP + pos) * 6;
      e[0] = x1; e[1] = y1; e[2] = x2; e[3] = y2;
      e[4] = 1.f / (1.f + __expf(-logit));
      e[5] = (float)lvl;
    }
  }
}

// ---------------- greedy NMS, grouped, one block per image ----------------
__global__ void k_nms(float* __restrict__ pool, const int* __restrict__ cnt, int cap,
                      float iouThr, int maxKeep, float* __restrict__ keepBox,
                      float* __restrict__ keepScore, float* __restrict__ keepGrp,
                      int* __restrict__ keepCnt) {
  int img = blockIdx.x;
  int n = cnt[img];
  if (n > cap) n = cap;
  float* e = pool + (size_t)img * cap * 6;
  __shared__ float sv[256];
  __shared__ int si[256];
  __shared__ float sbox[4];
  __shared__ int sgd[2];
  int kept = 0;
  for (int iter = 0; iter < n && kept < maxKeep; ++iter) {
    float bestv = -1e30f;
    int besti = -1;
    for (int j = threadIdx.x; j < n; j += 256) {
      float s = e[j * 6 + 4];
      if (s > bestv) { bestv = s; besti = j; }
    }
    sv[threadIdx.x] = bestv; si[threadIdx.x] = besti;
    __syncthreads();
    for (int o = 128; o > 0; o >>= 1) {
      if (threadIdx.x < o && sv[threadIdx.x + o] > sv[threadIdx.x]) {
        sv[threadIdx.x] = sv[threadIdx.x + o];
        si[threadIdx.x] = si[threadIdx.x + o];
      }
      __syncthreads();
    }
    if (threadIdx.x == 0) {
      int bi = si[0];
      if (bi < 0 || sv[0] <= -1e29f) {
        sgd[1] = 1;
      } else {
        sgd[1] = 0;
        float* b = e + bi * 6;
        sbox[0] = b[0]; sbox[1] = b[1]; sbox[2] = b[2]; sbox[3] = b[3];
        sgd[0] = (int)b[5];
        size_t ko = (size_t)(img * maxKeep + kept);
        keepBox[ko * 4 + 0] = b[0]; keepBox[ko * 4 + 1] = b[1];
        keepBox[ko * 4 + 2] = b[2]; keepBox[ko * 4 + 3] = b[3];
        keepScore[ko] = b[4];
        keepGrp[ko] = b[5];
        b[4] = -1e30f;
      }
    }
    __syncthreads();
    if (sgd[1]) break;
    float bx1 = sbox[0], by1 = sbox[1], bx2 = sbox[2], by2 = sbox[3];
    float barea = (bx2 - bx1) * (by2 - by1);
    int bg = sgd[0];
    for (int j = threadIdx.x; j < n; j += 256) {
      float s = e[j * 6 + 4];
      if (s <= -1e29f) continue;
      if ((int)e[j * 6 + 5] != bg) continue;
      float x1 = e[j * 6 + 0], y1 = e[j * 6 + 1], x2 = e[j * 6 + 2], y2 = e[j * 6 + 3];
      float iw = fminf(bx2, x2) - fmaxf(bx1, x1); if (iw < 0.f) iw = 0.f;
      float ih = fminf(by2, y2) - fmaxf(by1, y1); if (ih < 0.f) ih = 0.f;
      float inter = iw * ih;
      float iou = inter / (barea + (x2 - x1) * (y2 - y1) - inter + 1e-12f);
      if (iou > iouThr) e[j * 6 + 4] = -1e30f;
    }
    kept++;
    __syncthreads();
  }
  if (threadIdx.x == 0) keepCnt[img] = kept;
}

// ---------------- compact proposals from both images ----------------
__global__ void k_compact(const float* __restrict__ propBox, const int* __restrict__ propCnt,
                          float* __restrict__ RoiBox, int* __restrict__ RoiImg,
                          int* __restrict__ nRois) {
  int c0 = propCnt[0], c1 = propCnt[1];
  int tot = c0 + c1;
  if (tot > 1024) tot = 1024;
  for (int i = threadIdx.x; i < 1024; i += 256) {
    if (i < tot) {
      int img = (i < c0) ? 0 : 1;
      int k = (i < c0) ? i : i - c0;
      for (int j = 0; j < 4; ++j) RoiBox[i * 4 + j] = propBox[(img * 512 + k) * 4 + j];
      RoiImg[i] = img;
    } else {
      for (int j = 0; j < 4; ++j) RoiBox[i * 4 + j] = 0.f;
      RoiImg[i] = 0;
    }
  }
  if (threadIdx.x == 0) *nRois = tot;
}

// ---------------- multi-level ROI align -> bf16 features [roi][c*49+gy*7+gx] ---
__global__ void k_roialign(Ptrs5 fp, LvlInfo L, const float* __restrict__ RoiBox,
                           const int* __restrict__ RoiImg, const int* __restrict__ nRois,
                           unsigned short* __restrict__ Xroi) {
  int r = blockIdx.x;
  if (r >= *nRois) return;
  int cell = blockIdx.y;
  int gy = cell / ROI, gx = cell % ROI;
  int c = threadIdx.x;
  const float* bx = RoiBox + r * 4;
  float area = fmaxf((bx[2] - bx[0]) * (bx[3] - bx[1]), 1e-6f);
  float lv = floorf(4.0f + log2f(sqrtf(area) / 224.0f + 1e-6f));
  int lvl = (int)(fminf(fmaxf(lv, 2.0f), 5.0f)) - 2;
  float stride = (float)(4 << lvl);
  int Hh = L.Hh[lvl], Ww = L.Ww[lvl];
  int img = RoiImg[r];
  const float* fm = fp.p[lvl] + ((size_t)img * CCH + c) * Hh * Ww;
  float b0 = bx[0] / stride, b1 = bx[1] / stride, b2 = bx[2] / stride, b3 = bx[3] / stride;
  float rw = fmaxf(b2 - b0, 1.0f), rh = fmaxf(b3 - b1, 1.0f);
  float bw = rw / (float)ROI, bh = rh / (float)ROI;
  float acc = 0.f;
  for (int sy = 0; sy < 2; ++sy) {
    for (int sx = 0; sx < 2; ++sx) {
      float yy = b1 + ((float)gy + (sy + 0.5f) * 0.5f) * bh;
      float xx = b0 + ((float)gx + (sx + 0.5f) * 0.5f) * bw;
      bool valid = (yy > -1.0f) && (yy < (float)Hh) && (xx > -1.0f) && (xx < (float)Ww);
      if (valid) {
        float yc = fminf(fmaxf(yy, 0.f), (float)Hh - 1.f);
        float xc = fminf(fmaxf(xx, 0.f), (float)Ww - 1.f);
        int y0 = (int)floorf(yc); if (y0 < 0) y0 = 0; if (y0 > Hh - 1) y0 = Hh - 1;
        int x0 = (int)floorf(xc); if (x0 < 0) x0 = 0; if (x0 > Ww - 1) x0 = Ww - 1;
        int y1i = y0 + 1 < Hh ? y0 + 1 : Hh - 1;
        int x1i = x0 + 1 < Ww ? x0 + 1 : Ww - 1;
        float ly = yc - (float)y0, lx = xc - (float)x0;
        float v = fm[y0 * Ww + x0] * (1.f - ly) * (1.f - lx)
                + fm[y0 * Ww + x1i] * (1.f - ly) * lx
                + fm[y1i * Ww + x0] * ly * (1.f - lx)
                + fm[y1i * Ww + x1i] * ly * lx;
        acc += v;
      }
    }
  }
  acc *= 0.25f;
  Xroi[(size_t)r * KROI + c * (ROI * ROI) + gy * ROI + gx] = f2bf(acc);
}

// ---------------- class / box-reg heads + softmax ----------------
__global__ void k_cls_head(const unsigned short* __restrict__ Y, const float* __restrict__ clsw,
                           const float* __restrict__ clsb, const float* __restrict__ bregw,
                           const float* __restrict__ bregb, float* __restrict__ scores,
                           float* __restrict__ breg, const int* __restrict__ nRois) {
  int r = blockIdx.x;
  if (r >= *nRois) return;
  __shared__ float slog[NCLS + 1];
  __shared__ float smax, ssum;
  const unsigned short* xr = Y + (size_t)r * CLIN;
  const int NOUT = (NCLS + 1) + 4 * NCLS;
  for (int o = threadIdx.x; o < NOUT; o += blockDim.x) {
    float a;
    if (o < NCLS + 1) {
      a = clsb[o];
      for (int k = 0; k < CLIN; ++k) a += bf2f(xr[k]) * clsw[k * (NCLS + 1) + o];
      slog[o] = a;
    } else {
      int j = o - (NCLS + 1);
      a = bregb[j];
      for (int k = 0; k < CLIN; ++k) a += bf2f(xr[k]) * bregw[k * (4 * NCLS) + j];
      breg[(size_t)r * 4 * NCLS + j] = a;
    }
  }
  __syncthreads();
  if (threadIdx.x == 0) {
    float mx = slog[0];
    for (int i = 1; i < NCLS + 1; ++i) mx = fmaxf(mx, slog[i]);
    float s = 0.f;
    for (int i = 0; i < NCLS + 1; ++i) s += __expf(slog[i] - mx);
    smax = mx; ssum = s;
  }
  __syncthreads();
  for (int cc = threadIdx.x + 1; cc < NCLS + 1; cc += blockDim.x)
    scores[(size_t)r * NCLS + (cc - 1)] = __expf(slog[cc] - smax) / ssum;
}

// ---------------- stage-2 decode + filter -> per-image pool ----------------
__global__ void k_stage2(const float* __restrict__ scores, const float* __restrict__ breg,
                         const float* __restrict__ RoiBox, const int* __restrict__ RoiImg,
                         const float* __restrict__ imsizes, float* __restrict__ pool2,
                         int* __restrict__ cnt2, const int* __restrict__ nRois) {
  int id = blockIdx.x * blockDim.x + threadIdx.x;
  int total = (*nRois) * NCLS;
  if (id >= total) return;
  int r = id / NCLS, c = id % NCLS;
  float sc = scores[(size_t)r * NCLS + c];
  if (sc <= 0.05f) return;
  const float* pb = RoiBox + r * 4;
  float pcx = (pb[0] + pb[2]) * 0.5f, pcy = (pb[1] + pb[3]) * 0.5f;
  float pw = pb[2] - pb[0], ph = pb[3] - pb[1];
  const float* rg = breg + (size_t)r * 4 * NCLS + c * 4;
  float cx = pcx + rg[0] * 0.1f * pw;
  float cy = pcy + rg[1] * 0.1f * ph;
  float bw = pw * __expf(fminf(rg[2] * 0.2f, CLIPV));
  float bh = ph * __expf(fminf(rg[3] * 0.2f, CLIPV));
  int img = RoiImg[r];
  float Wi = imsizes[img * 2 + 1], Hi = imsizes[img * 2 + 0];
  float x1 = fminf(fmaxf(cx - bw * 0.5f, 0.f), Wi);
  float x2 = fminf(fmaxf(cx + bw * 0.5f, 0.f), Wi);
  float y1 = fminf(fmaxf(cy - bh * 0.5f, 0.f), Hi);
  float y2 = fminf(fmaxf(cy + bh * 0.5f, 0.f), Hi);
  if ((x2 - x1) >= 0.01f && (y2 - y1) >= 0.01f) {
    int pos = atomicAdd(&cnt2[img], 1);
    if (pos < P2CAP) {
      float* e = pool2 + ((size_t)img * P2CAP + pos) * 6;
      e[0] = x1; e[1] = y1; e[2] = x2; e[3] = y2;
      e[4] = sc;
      e[5] = (float)(c + 1);
    }
  }
}

// ---------------- final output concatenation ----------------
__global__ void k_write_out(const float* __restrict__ fb, const float* __restrict__ fs,
                            const float* __restrict__ fc, const int* __restrict__ outCnt,
                            float* __restrict__ out, int outSize) {
  int c0 = outCnt[0], c1 = outCnt[1];
  int R = c0 + c1;
  for (int i = threadIdx.x; i < R; i += blockDim.x) {
    int img = (i < c0) ? 0 : 1;
    int k = (i < c0) ? i : i - c0;
    for (int j = 0; j < 4; ++j) {
      int o = i * 4 + j;
      if (o < outSize) out[o] = fb[(img * 100 + k) * 4 + j];
    }
    int o1 = 4 * R + i; if (o1 < outSize) out[o1] = fs[img * 100 + k];
    int o2 = 5 * R + i; if (o2 < outSize) out[o2] = fc[img * 100 + k];
    int o3 = 6 * R + i; if (o3 < outSize) out[o3] = (float)img;
  }
}

// ================================================================================
extern "C" void kernel_launch(void* const* d_in, const int* in_sizes, int n_in,
                              void* d_out, int out_size, void* d_ws, size_t ws_size,
                              hipStream_t stream) {
  (void)in_sizes; (void)n_in; (void)ws_size;
  const float* fmap[NLVL];
  const float* prior[NLVL];
  for (int l = 0; l < NLVL; ++l) fmap[l] = (const float*)d_in[l];
  for (int l = 0; l < NLVL; ++l) prior[l] = (const float*)d_in[5 + l];
  const float* imsizes = (const float*)d_in[10];
  const float* conv_w = (const float*)d_in[11];
  const float* conv_b = (const float*)d_in[12];
  const float* log_w  = (const float*)d_in[13];
  const float* log_b  = (const float*)d_in[14];
  const float* reg_w  = (const float*)d_in[15];
  const float* reg_b  = (const float*)d_in[16];
  const float* fc1_w  = (const float*)d_in[17];
  const float* fc1_b  = (const float*)d_in[18];
  const float* fc2_w  = (const float*)d_in[19];
  const float* fc2_b  = (const float*)d_in[20];
  const float* cls_w  = (const float*)d_in[21];
  const float* cls_b  = (const float*)d_in[22];
  const float* breg_w = (const float*)d_in[23];
  const float* breg_b = (const float*)d_in[24];

  const int LHh[NLVL] = {200, 100, 50, 25, 13};
  const int LWw[NLVL] = {304, 152, 76, 38, 19};
  LvlInfo L;
  {
    int pOff = 0, fOff = 0, padOff = 0;
    for (int l = 0; l < NLVL; ++l) {
      L.Hh[l] = LHh[l]; L.Ww[l] = LWw[l];
      L.pixOff[l] = pOff; L.padOff[l] = padOff; L.featOff[l] = fOff;
      int P = LHh[l] * LWw[l];
      L.Fl[l] = P * AANCH;
      pOff += P;
      padOff += (LHh[l] + 2) * (LWw[l] + 2);
      fOff += P * AANCH;
    }
  }

  char* ws = (char*)d_ws;
  size_t cur = 0;
  auto nb = [&](size_t bytes) -> char* {
    char* p = ws + cur;
    cur = (cur + bytes + 255) & ~(size_t)255;
    return p;
  };
  unsigned short* X    = (unsigned short*)nb((size_t)NIMG * PTP * CCH * 2);
  unsigned short* Hf   = (unsigned short*)nb((size_t)NIMG * PT * CCH * 2);
  unsigned short* Wp   = (unsigned short*)nb((size_t)CCH * K9 * 2);
  unsigned short* W1t  = (unsigned short*)nb((size_t)CLIN * KROI * 2);
  unsigned short* W2t  = (unsigned short*)nb((size_t)CLIN * CLIN * 2);
  float* Lg            = (float*)nb((size_t)NIMG * FT * 4);
  float* Rg            = (float*)nb((size_t)NIMG * FT * 16);
  float* candScore     = (float*)nb((size_t)NIMG * NLVL * TOPK_CAP * 4);
  int*   candIdx       = (int*)nb((size_t)NIMG * NLVL * TOPK_CAP * 4);
  float* pool1         = (float*)nb((size_t)NIMG * POOLCAP * 6 * 4);
  float* propBox       = (float*)nb((size_t)NIMG * 512 * 4 * 4);
  float* propScoreD    = (float*)nb((size_t)NIMG * 512 * 4);
  float* propGrpD      = (float*)nb((size_t)NIMG * 512 * 4);
  float* RoiBox        = (float*)nb((size_t)1024 * 4 * 4);
  int*   RoiImg        = (int*)nb((size_t)1024 * 4);
  unsigned short* Xroi = (unsigned short*)nb((size_t)1024 * KROI * 2);
  unsigned short* Y1   = (unsigned short*)nb((size_t)1024 * CLIN * 2);
  unsigned short* Y2   = (unsigned short*)nb((size_t)1024 * CLIN * 2);
  float* S2s           = (float*)nb((size_t)1024 * NCLS * 4);
  float* S2r           = (float*)nb((size_t)1024 * 4 * NCLS * 4);
  float* pool2         = (float*)nb((size_t)NIMG * P2CAP * 6 * 4);
  float* finalBox      = (float*)nb((size_t)NIMG * 100 * 4 * 4);
  float* finalScore    = (float*)nb((size_t)NIMG * 100 * 4);
  float* finalCls      = (float*)nb((size_t)NIMG * 100 * 4);
  int*   counters      = (int*)nb(64 * 4);
  int* candCnt    = counters;       // 10
  int* imgCandCnt = counters + 10;  // 2
  int* propCnt    = counters + 12;  // 2
  int* nRois      = counters + 14;  // 1
  int* cnt2       = counters + 15;  // 2
  int* outCnt     = counters + 17;  // 2

  Ptrs5 fm5, pr5, fm4;
  for (int l = 0; l < NLVL; ++l) { fm5.p[l] = fmap[l]; pr5.p[l] = prior[l]; }
  for (int l = 0; l < NLVL; ++l) fm4.p[l] = fmap[l < 4 ? l : 3];

  // 0) zero counters, padded X (halo), ROI features (padding rows), d_out
  k_zero_u32<<<1, 64, 0, stream>>>((unsigned*)counters, 64);
  k_zero_u32<<<2048, 256, 0, stream>>>((unsigned*)X, (size_t)NIMG * PTP * CCH / 2);
  k_zero_u32<<<1024, 256, 0, stream>>>((unsigned*)Xroi, (size_t)1024 * KROI / 2);
  k_zero_u32<<<256, 256, 0, stream>>>((unsigned*)d_out, (size_t)out_size);

  // 1) pack weights + activations to bf16 WMMA-friendly layouts
  {
    int tot = CCH * K9;
    k_pack_conv_w<<<(tot + 255) / 256, 256, 0, stream>>>(conv_w, Wp);
  }
  {
    size_t tot = (size_t)KROI * CLIN;
    k_pack_fcw<<<(unsigned)((tot + 255) / 256), 256, 0, stream>>>(fc1_w, W1t, KROI, CLIN);
  }
  {
    size_t tot = (size_t)CLIN * CLIN;
    k_pack_fcw<<<(unsigned)((tot + 255) / 256), 256, 0, stream>>>(fc2_w, W2t, CLIN, CLIN);
  }
  {
    size_t tot = (size_t)NIMG * PT * CCH;
    k_pack_x_pad<<<(unsigned)((tot + 255) / 256), 256, 0, stream>>>(fm5, L, X);
  }

  // 2) conv3x3 + ReLU per level (bf16 WMMA implicit GEMM, async-LDS staged A)
  for (int l = 0; l < NLVL; ++l) {
    int P = LHh[l] * LWw[l];
    dim3 g((P + 15) / 16, 1, NIMG);
    k_conv3x3_wmma<<<g, 256, 0, stream>>>(X, Wp, conv_b, Hf,
                                          LHh[l], LWw[l], L.pixOff[l], L.padOff[l]);
  }

  // 3) RPN heads
  k_heads<<<(NIMG * PT + 255) / 256, 256, 0, stream>>>(Hf, log_w, log_b, reg_w, reg_b, Lg, Rg, L);

  // 4) top-k, decode, per-image NMS -> proposals
  k_topk<<<NIMG * NLVL, 256, 0, stream>>>(Lg, L, candScore, candIdx, candCnt);
  k_decode<<<(NIMG * NLVL * TOPK_CAP + 255) / 256, 256, 0, stream>>>(
      candScore, candIdx, candCnt, Rg, pr5, imsizes, pool1, imgCandCnt, L);
  k_nms<<<NIMG, 256, 0, stream>>>(pool1, imgCandCnt, POOLCAP, 0.7f, 512,
                                  propBox, propScoreD, propGrpD, propCnt);
  k_compact<<<1, 256, 0, stream>>>(propBox, propCnt, RoiBox, RoiImg, nRois);

  // 5) multi-level ROI align -> bf16 feature matrix [1024 x 12544]
  {
    dim3 g(1024, ROI * ROI);
    k_roialign<<<g, CCH, 0, stream>>>(fm4, L, RoiBox, RoiImg, nRois, Xroi);
  }

  // 6) FC1/FC2 as bf16 WMMA GEMMs with async-LDS staged A
  {
    dim3 g(1024 / 16, CLIN / 256);
    k_gemm_wmma<<<g, 256, 0, stream>>>(Xroi, W1t, fc1_b, Y1, 1024, KROI, CLIN, 1);
    k_gemm_wmma<<<g, 256, 0, stream>>>(Y1, W2t, fc2_b, Y2, 1024, CLIN, CLIN, 1);
  }

  // 7) heads + stage-2 decode + class-grouped NMS
  k_cls_head<<<1024, 128, 0, stream>>>(Y2, cls_w, cls_b, breg_w, breg_b, S2s, S2r, nRois);
  k_stage2<<<(1024 * NCLS + 255) / 256, 256, 0, stream>>>(S2s, S2r, RoiBox, RoiImg, imsizes,
                                                          pool2, cnt2, nRois);
  k_nms<<<NIMG, 256, 0, stream>>>(pool2, cnt2, P2CAP, 0.5f, 100,
                                  finalBox, finalScore, finalCls, outCnt);

  // 8) concatenate outputs
  k_write_out<<<1, 256, 0, stream>>>(finalBox, finalScore, finalCls, outCnt,
                                     (float*)d_out, out_size);
}
